// MyAttention_81905026335414
// MI455X (gfx1250) — compile-verified, matrix operand order checked by reading, not run on hardware
//
#include <hip/hip_runtime.h>

// ---------------- types & helpers ----------------
typedef __attribute__((ext_vector_type(16))) __bf16 v16bf;
typedef __attribute__((ext_vector_type(2)))  __bf16 v2bf;
typedef __attribute__((ext_vector_type(8)))  float  v8f;

#define WMMA_BF16(a, b, c) \
  __builtin_amdgcn_wmma_f32_16x16x32_bf16(false, (a), false, (b), (short)0, (c), false, false)

// native fptrunc: gfx1250 has hardware f32->bf16 converts (RNE)
static __device__ __forceinline__ __bf16 f2bf(float f) { return (__bf16)f; }

// pack two fp32 -> packed bf16x2 dword (lo = a, hi = b) via native converts
static __device__ __forceinline__ unsigned pack2bf(float a, float b) {
  v2bf v;
  v[0] = (__bf16)a;
  v[1] = (__bf16)b;
  return __builtin_bit_cast(unsigned, v);
}

// K-pair base for 16-bit A/B fragment element (VGPR pair p, per ISA 7.12.2)
static __device__ __forceinline__ int kpair(int p, int lane) {
  int base = (p < 4) ? (2 * p) : (16 + 2 * (p - 4));
  return base + ((lane & 16) ? 8 : 0);
}

// Gather a 16-lane-rows x 32-K bf16 fragment from an LDS tile (row stride ld)
static __device__ __forceinline__ v16bf load_frag(const __bf16* buf, int rowBase, int ld,
                                                  int kOff, int lane) {
  v16bf f;
  int r = rowBase + (lane & 15);
#pragma unroll
  for (int p = 0; p < 8; ++p) {
    int k = kOff + kpair(p, lane);
    f[2 * p]     = buf[r * ld + k];
    f[2 * p + 1] = buf[r * ld + k + 1];
  }
  return f;
}

// CDNA5 async global->LDS copy, 16 bytes per lane (ASYNCcnt-tracked, no VGPR data)
static __device__ __forceinline__ void async_b128(const void* lds_dst, const void* gsrc) {
  unsigned l = (unsigned)(unsigned long long)lds_dst;     // wave-relative LDS byte addr
  unsigned long long g = (unsigned long long)gsrc;
  asm volatile("global_load_async_to_lds_b128 %0, %1, off" :: "v"(l), "v"(g) : "memory");
}
static __device__ __forceinline__ void wait_async0() {
  asm volatile("s_wait_asynccnt 0x0" ::: "memory");
}
// leave the 2 just-issued prefetch copies in flight; older ones drain in order
static __device__ __forceinline__ void wait_async2() {
  asm volatile("s_wait_asynccnt 0x2" ::: "memory");
}

// ---------------- problem constants ----------------
#define BB    2
#define NQ    4096
#define NSRC  3072
#define CC    512
#define HH    32
#define WW    32
#define HWH   1024          // H*W
#define NHEAD 8
#define HD    64

// ---------------- kernel 1: zero accumulators ----------------
__global__ void zero_kernel(float* p, int n) {
  int i = blockIdx.x * blockDim.x + threadIdx.x;
  if (i < n) p[i] = 0.0f;
}

// ---------------- kernel 2: token2map scatter ----------------
__global__ void scatter_kernel(const float* __restrict__ xs, const float* __restrict__ loc,
                               const float* __restrict__ conf,
                               float* __restrict__ feat, float* __restrict__ cnt,
                               float* __restrict__ cacc) {
  int token = blockIdx.x;                 // 0 .. B*Nsrc-1
  int b = token / NSRC;
  float lx = loc[token * 2 + 0];
  float ly = loc[token * 2 + 1];
  lx = fminf(fmaxf(lx, -1.0f), 1.0f);
  ly = fminf(fmaxf(ly, -1.0f), 1.0f);
  int px = (int)floorf(0.5f * (lx + 1.0f) * WW - 0.5f + 0.5f);
  int py = (int)floorf(0.5f * (ly + 1.0f) * HH - 0.5f + 0.5f);
  px = min(max(px, 0), WW - 1);
  py = min(max(py, 0), HH - 1);
  int idx = b * HWH + py * WW + px;
  for (int c = threadIdx.x; c < CC; c += blockDim.x)
    atomicAdd(&feat[(size_t)idx * CC + c], xs[(size_t)token * CC + c]);
  if (threadIdx.x == 0) {
    atomicAdd(&cnt[idx], 1.0f);
    atomicAdd(&cacc[idx], conf[token]);
  }
}

// ---------------- kernel 3: normalize maps ----------------
__global__ void normalize_kernel(const float* __restrict__ feat, const float* __restrict__ cnt,
                                 const float* __restrict__ cacc,
                                 float* __restrict__ srcmap, float* __restrict__ confmap) {
  int cell = blockIdx.x;                  // 0 .. B*HW-1
  float inv = 1.0f / (cnt[cell] + 1e-6f);
  for (int c = threadIdx.x; c < CC; c += blockDim.x)
    srcmap[(size_t)cell * CC + c] = feat[(size_t)cell * CC + c] * inv;
  if (threadIdx.x == 0) confmap[cell] = cacc[cell] * inv;
}

// ---------------- kernel 4: WMMA GEMM  out = A @ W^T (+bias) ----------------
// A: [M,K] fp32 row-major, W: [N,K] fp32 row-major. 64x128 tile, K-step 32, 8 waves,
// software-pipelined staging: next tile's float4 loads issue before current compute.
__global__ __launch_bounds__(256) void gemm_xwT(const float* __restrict__ A,
                                                const float* __restrict__ Wt,
                                                const float* __restrict__ bias,
                                                float* __restrict__ outF,
                                                __bf16* __restrict__ outB,
                                                int M, int N, int K) {
  __shared__ __align__(16) unsigned As32[64 * 16];    // 64x32 bf16 (packed pairs)
  __shared__ __align__(16) unsigned Bs32[128 * 16];   // 128x32 bf16
  const __bf16* As = (const __bf16*)As32;
  const __bf16* Bs = (const __bf16*)Bs32;
  int tid = threadIdx.x, lane = tid & 31, w = tid >> 5;
  int wm = w & 3;          // 4 row-groups of 16
  int wn = w >> 2;         // 2 col-groups of 64
  int rowBase = blockIdx.y * 64;
  int colBase = blockIdx.x * 128;
  int r2 = tid >> 3, c4 = (tid & 7) * 4;   // staging coords
  v8f zero = {};
  v8f acc[4];
#pragma unroll
  for (int t = 0; t < 4; ++t) acc[t] = zero;

  float4 ra0, ra1, rb0, rb1, rb2, rb3;
  auto loadG = [&](int k0) {
    ra0 = *(const float4*)&A[(size_t)(rowBase + r2) * K + k0 + c4];
    ra1 = *(const float4*)&A[(size_t)(rowBase + r2 + 32) * K + k0 + c4];
    rb0 = *(const float4*)&Wt[(size_t)(colBase + r2) * K + k0 + c4];
    rb1 = *(const float4*)&Wt[(size_t)(colBase + r2 + 32) * K + k0 + c4];
    rb2 = *(const float4*)&Wt[(size_t)(colBase + r2 + 64) * K + k0 + c4];
    rb3 = *(const float4*)&Wt[(size_t)(colBase + r2 + 96) * K + k0 + c4];
  };
  auto storeL = [&]() {
    unsigned a0 = (unsigned)(r2 * 32 + c4) >> 1;
    unsigned a1 = (unsigned)((r2 + 32) * 32 + c4) >> 1;
    As32[a0] = pack2bf(ra0.x, ra0.y); As32[a0 + 1] = pack2bf(ra0.z, ra0.w);
    As32[a1] = pack2bf(ra1.x, ra1.y); As32[a1 + 1] = pack2bf(ra1.z, ra1.w);
    unsigned b0 = (unsigned)(r2 * 32 + c4) >> 1;
    unsigned b1 = (unsigned)((r2 + 32) * 32 + c4) >> 1;
    unsigned b2 = (unsigned)((r2 + 64) * 32 + c4) >> 1;
    unsigned b3 = (unsigned)((r2 + 96) * 32 + c4) >> 1;
    Bs32[b0] = pack2bf(rb0.x, rb0.y); Bs32[b0 + 1] = pack2bf(rb0.z, rb0.w);
    Bs32[b1] = pack2bf(rb1.x, rb1.y); Bs32[b1 + 1] = pack2bf(rb1.z, rb1.w);
    Bs32[b2] = pack2bf(rb2.x, rb2.y); Bs32[b2 + 1] = pack2bf(rb2.z, rb2.w);
    Bs32[b3] = pack2bf(rb3.x, rb3.y); Bs32[b3 + 1] = pack2bf(rb3.z, rb3.w);
  };

  loadG(0);
  storeL();
  __syncthreads();
  for (int k0 = 0; k0 < K; k0 += 32) {
    bool more = (k0 + 32 < K);
    if (more) loadG(k0 + 32);                       // overlap with compute below
    if (k0 + 64 < K) {
      __builtin_prefetch(&A[(size_t)(rowBase + r2) * K + k0 + 64 + c4], 0, 0);
      __builtin_prefetch(&Wt[(size_t)(colBase + r2) * K + k0 + 64 + c4], 0, 0);
    }
    v16bf a = load_frag(As, wm * 16, 32, 0, lane);
#pragma unroll
    for (int t = 0; t < 4; ++t) {
      v16bf bf = load_frag(Bs, wn * 64 + t * 16, 32, 0, lane);
      acc[t] = WMMA_BF16(a, bf, acc[t]);
    }
    __syncthreads();
    if (more) {
      storeL();
      __syncthreads();
    }
  }
  int halfSel = lane >> 4;
  int colL = lane & 15;
#pragma unroll
  for (int t = 0; t < 4; ++t) {
    int col = colBase + wn * 64 + t * 16 + colL;
    float bv = bias ? bias[col] : 0.0f;
#pragma unroll
    for (int r = 0; r < 8; ++r) {
      int row = rowBase + wm * 16 + r + halfSel * 8;
      float v = acc[t][r] + bv;
      if (outF) outF[(size_t)row * N + col] = v;
      if (outB) outB[(size_t)row * N + col] = f2bf(v);
    }
  }
}

// ---------------- kernel 5: LayerNorm + exact GELU ----------------
__global__ __launch_bounds__(256) void ln_gelu_kernel(const float* __restrict__ in,
                                                      const float* __restrict__ g,
                                                      const float* __restrict__ be,
                                                      float* __restrict__ out) {
  int row = blockIdx.x;
  int tid = threadIdx.x;
  float v0 = in[(size_t)row * CC + tid];
  float v1 = in[(size_t)row * CC + tid + 256];
  float s = v0 + v1, s2 = v0 * v0 + v1 * v1;
#pragma unroll
  for (int off = 16; off > 0; off >>= 1) {
    s  += __shfl_xor(s,  off, 32);
    s2 += __shfl_xor(s2, off, 32);
  }
  __shared__ float ps[8], ps2[8];
  int w = tid >> 5, lane = tid & 31;
  if (lane == 0) { ps[w] = s; ps2[w] = s2; }
  __syncthreads();
  __shared__ float muS, varS;
  if (tid == 0) {
    float a = 0.f, b2 = 0.f;
    for (int i = 0; i < 8; ++i) { a += ps[i]; b2 += ps2[i]; }
    float mu = a / CC;
    muS = mu;
    varS = b2 / CC - mu * mu;
  }
  __syncthreads();
  float mu = muS;
  float rstd = rsqrtf(varS + 1e-5f);
  float y0 = (v0 - mu) * rstd * g[tid] + be[tid];
  float y1 = (v1 - mu) * rstd * g[tid + 256] + be[tid + 256];
  out[(size_t)row * CC + tid]       = 0.5f * y0 * (1.0f + erff(y0 * 0.70710678118f));
  out[(size_t)row * CC + tid + 256] = 0.5f * y1 * (1.0f + erff(y1 * 0.70710678118f));
}

// ---------------- kernel 6: fused attention ----------------
// One block = (b, h, 32 query rows). Full 32x1024 score tile kept in LDS (128 KB).
// k/v tiles are double-buffered with async global->LDS copies: the next tile's DMA
// is in flight (ASYNCcnt==2) while the current tile feeds the WMMAs.
__global__ __launch_bounds__(256) void attn_kernel(const __bf16* __restrict__ qb,
                                                   const __bf16* __restrict__ kvb,
                                                   const float* __restrict__ confmap,
                                                   float* __restrict__ attn_out,
                                                   float scale) {
  __shared__ __align__(16) __bf16 Qs[32 * 64];
  __shared__ __align__(16) __bf16 KV0[64 * 64];
  __shared__ __align__(16) __bf16 KV1[64 * 64];
  __shared__ float S[32 * HWH];         // 128 KB
  const int Ckv = 2 * CC;
  int bh = blockIdx.y;                  // 0..15
  int b = bh >> 3, h = bh & 7;
  int m0 = blockIdx.x * 32;
  int tid = threadIdx.x, lane = tid & 31, w = tid >> 5;
  int wm = w & 1;                       // 2 row-groups of 16
  int wn = w >> 1;                      // 4 col-groups of 16

  // issue 2 async 16B copies per thread for one 64x64 bf16 tile
  auto stage_kv = [&](int chunk, __bf16* dst, int colOff) {
#pragma unroll
    for (int i = tid; i < 512; i += 256) {
      int r = i >> 3, c8 = (i & 7) * 8;
      async_b128(&dst[r * 64 + c8],
                 &kvb[(size_t)(b * HWH + chunk * 64 + r) * Ckv + colOff + c8]);
    }
  };

  // prologue: q chunk (1 copy/thread) + k chunk 0
  {
    int r = tid >> 3, c8 = (tid & 7) * 8;
    async_b128(&Qs[r * 64 + c8], &qb[(size_t)(b * NQ + m0 + r) * CC + h * HD + c8]);
  }
  stage_kv(0, KV0, h * HD);

  // phase 1: S = q k^T * scale + conf-bias (double-buffered k tiles)
  for (int it = 0; it < 16; ++it) {
    __bf16* cur = (it & 1) ? KV1 : KV0;
    __bf16* nxt = (it & 1) ? KV0 : KV1;
    if (it + 1 < 16) { stage_kv(it + 1, nxt, h * HD); wait_async2(); }
    else             { wait_async0(); }
    __syncthreads();
    v16bf a0 = load_frag(Qs, wm * 16, 64, 0, lane);
    v16bf a1 = load_frag(Qs, wm * 16, 64, 32, lane);
    v16bf b0 = load_frag(cur, wn * 16, 64, 0, lane);
    v16bf b1 = load_frag(cur, wn * 16, 64, 32, lane);
    v8f acc = {};
    acc = WMMA_BF16(a0, b0, acc);
    acc = WMMA_BF16(a1, b1, acc);
    int col = it * 64 + wn * 16 + (lane & 15);
    float bias = confmap[b * HWH + col];
#pragma unroll
    for (int r = 0; r < 8; ++r) {
      int row = wm * 16 + r + (lane >> 4) * 8;
      S[row * HWH + col] = acc[r] * scale + bias;
    }
    __syncthreads();
  }

  // phase 2: softmax over the 1024 columns, 4 rows per wave
  for (int r = w * 4; r < w * 4 + 4; ++r) {
    float m = -3.4e38f;
    for (int c = lane; c < HWH; c += 32) m = fmaxf(m, S[r * HWH + c]);
#pragma unroll
    for (int off = 16; off > 0; off >>= 1) m = fmaxf(m, __shfl_xor(m, off, 32));
    float sum = 0.f;
    for (int c = lane; c < HWH; c += 32) {
      float e = __expf(S[r * HWH + c] - m);
      S[r * HWH + c] = e;
      sum += e;
    }
#pragma unroll
    for (int off = 16; off > 0; off >>= 1) sum += __shfl_xor(sum, off, 32);
    float inv = 1.0f / sum;
    for (int c = lane; c < HWH; c += 32) S[r * HWH + c] *= inv;
  }
  __syncthreads();

  // phase 3: O = P @ v (double-buffered v tiles)
  v8f o = {};
  stage_kv(0, KV0, CC + h * HD);
  for (int it = 0; it < 16; ++it) {
    __bf16* cur = (it & 1) ? KV1 : KV0;
    __bf16* nxt = (it & 1) ? KV0 : KV1;
    if (it + 1 < 16) { stage_kv(it + 1, nxt, CC + h * HD); wait_async2(); }
    else             { wait_async0(); }
    __syncthreads();
    int kk = it * 64;
#pragma unroll
    for (int step = 0; step < 2; ++step) {
      v16bf a;
      int ar = wm * 16 + (lane & 15);
#pragma unroll
      for (int p = 0; p < 8; ++p) {
        int k = kk + step * 32 + kpair(p, lane);
        a[2 * p]     = f2bf(S[ar * HWH + k]);
        a[2 * p + 1] = f2bf(S[ar * HWH + k + 1]);
      }
      v16bf bb;
      int nc = wn * 16 + (lane & 15);
#pragma unroll
      for (int p = 0; p < 8; ++p) {
        int k = step * 32 + kpair(p, lane);
        bb[2 * p]     = cur[k * 64 + nc];
        bb[2 * p + 1] = cur[(k + 1) * 64 + nc];
      }
      o = WMMA_BF16(a, bb, o);
    }
    __syncthreads();
  }
  int colo = h * HD + wn * 16 + (lane & 15);
#pragma unroll
  for (int r = 0; r < 8; ++r) {
    int row = m0 + wm * 16 + r + (lane >> 4) * 8;
    attn_out[(size_t)(b * NQ + row) * CC + colo] = o[r];
  }
}

// ---------------- host launcher ----------------
extern "C" void kernel_launch(void* const* d_in, const int* in_sizes, int n_in,
                              void* d_out, int out_size, void* d_ws, size_t ws_size,
                              hipStream_t stream) {
  (void)in_sizes; (void)n_in; (void)out_size; (void)ws_size;
  const float* x     = (const float*)d_in[0];
  const float* xs    = (const float*)d_in[1];
  const float* loc   = (const float*)d_in[2];
  const float* conf  = (const float*)d_in[3];
  const float* Wq    = (const float*)d_in[4];
  const float* Wkv   = (const float*)d_in[5];
  const float* Wsr   = (const float*)d_in[6];
  const float* bsr   = (const float*)d_in[7];
  const float* gamma = (const float*)d_in[8];
  const float* beta  = (const float*)d_in[9];
  const float* Wproj = (const float*)d_in[10];
  const float* bproj = (const float*)d_in[11];
  float* out = (float*)d_out;

  char* ws = (char*)d_ws;
  size_t off = 0;
  auto alloc = [&](size_t bytes) -> void* {
    void* p = ws + off;
    off += (bytes + 255) & ~(size_t)255;
    return p;
  };
  float*  feat    = (float*) alloc((size_t)BB * HWH * CC * 4);   // scatter accum
  float*  cnt     = (float*) alloc((size_t)BB * HWH * 4);
  float*  cacc    = (float*) alloc((size_t)BB * HWH * 4);
  float*  confmap = (float*) alloc((size_t)BB * HWH * 4);
  float*  srcmap  = (float*) alloc((size_t)BB * HWH * CC * 4);
  float*  srlin   = (float*) alloc((size_t)BB * HWH * CC * 4);
  float*  srcact  = (float*) alloc((size_t)BB * HWH * CC * 4);
  __bf16* q_bf    = (__bf16*)alloc((size_t)BB * NQ * CC * 2);
  __bf16* kv_bf   = (__bf16*)alloc((size_t)BB * HWH * 2 * CC * 2);
  float*  attn_o  = (float*) alloc((size_t)BB * NQ * CC * 4);

  // 1. zero scatter accumulators (feat+cnt+cacc contiguous)
  int nz = BB * HWH * CC + 2 * BB * HWH;
  zero_kernel<<<(nz + 255) / 256, 256, 0, stream>>>(feat, nz);

  // 2. token2map scatter
  scatter_kernel<<<BB * NSRC, 128, 0, stream>>>(xs, loc, conf, feat, cnt, cacc);

  // 3. normalize
  normalize_kernel<<<BB * HWH, 128, 0, stream>>>(feat, cnt, cacc, srcmap, confmap);

  // 4. sr GEMM: srcmap @ Wsr^T + bsr  [2048,512]x[512,512]
  gemm_xwT<<<dim3(CC / 128, BB * HWH / 64), 256, 0, stream>>>(
      srcmap, Wsr, bsr, srlin, (__bf16*)nullptr, BB * HWH, CC, CC);

  // 5. LN + GELU
  ln_gelu_kernel<<<BB * HWH, 256, 0, stream>>>(srlin, gamma, beta, srcact);

  // 6. q GEMM: x @ Wq^T -> bf16  [8192,512]x[512,512]
  gemm_xwT<<<dim3(CC / 128, BB * NQ / 64), 256, 0, stream>>>(
      x, Wq, (const float*)nullptr, (float*)nullptr, q_bf, BB * NQ, CC, CC);

  // 7. kv GEMM: srcact @ Wkv^T -> bf16  [2048,512]x[1024,512]
  gemm_xwT<<<dim3(2 * CC / 128, BB * HWH / 64), 256, 0, stream>>>(
      srcact, Wkv, (const float*)nullptr, (float*)nullptr, kv_bf, BB * HWH, 2 * CC, CC);

  // 8. fused attention
  attn_kernel<<<dim3(NQ / 32, BB * NHEAD), 256, 0, stream>>>(
      q_bf, kv_bf, confmap, attn_o, 0.125f);

  // 9. output projection: attn_o @ Wproj^T + bproj -> d_out
  gemm_xwT<<<dim3(CC / 128, BB * NQ / 64), 256, 0, stream>>>(
      attn_o, Wproj, bproj, out, (__bf16*)nullptr, BB * NQ, CC, CC);
}
